// Model_61469571940899
// MI455X (gfx1250) — compile-verified, matrix operand order checked by reading, not run on hardware
//
#include <hip/hip_runtime.h>
#include <hip/hip_bf16.h>

typedef __attribute__((ext_vector_type(16))) __bf16 v16bf;
typedef __attribute__((ext_vector_type(8)))  __bf16 v8bf;
typedef __attribute__((ext_vector_type(8)))  float  v8f;

// exact pointee type from hipcc's diagnostic for the async-LDS builtin
typedef __attribute__((__vector_size__(4 * sizeof(int)))) int v4i_t;
typedef __attribute__((address_space(1))) v4i_t* as1_v4i;
typedef __attribute__((address_space(3))) v4i_t* as3_v4i;

#define BB 32
#define NN 64
#define LL 64
#define TT 10
#define DD 256
#define HH 512
#define H3 (3 * HH)

#if defined(__has_builtin)
#if __has_builtin(__builtin_amdgcn_global_load_async_to_lds_b128)
#define HAVE_ASYNC_LDS 1
#endif
#if __has_builtin(__builtin_amdgcn_s_wait_asynccnt)
#define HAVE_WAIT_ASYNC 1
#endif
#endif

__device__ __forceinline__ void wait_asynccnt0() {
#if defined(HAVE_WAIT_ASYNC)
  __builtin_amdgcn_s_wait_asynccnt(0);
#else
  asm volatile("s_wait_asynccnt 0x0" ::: "memory");
#endif
}

// ---------------------------------------------------------------------------
// Core: one wave computes a 16x64 tile of C = A @ B^T' where
//   A  : bf16 row-major [M,K]   (k-contiguous), stride sA
//   BT : bf16 row-major [N,K]   (k-contiguous), stride sB  (i.e. B transposed)
// 4 independent f32 accumulators; A fragment loaded once per K-chunk.
// Per ISA 7.12.2: lane&15 owns A-row/B-col; lanes>=16 hold the +8/+24 K-halves.
// ---------------------------------------------------------------------------
__device__ __forceinline__ void wmma_tile16x64(v8f acc[4],
    const __bf16* __restrict__ A, int sA,
    const __bf16* __restrict__ BT, int sB,
    int K, int lane)
{
  const int mr   = lane & 15;
  const int koff = (lane >> 4) * 8;
  const __bf16* Ar = A + (size_t)mr * sA;
  for (int kk = 0; kk < K; kk += 32) {
    v8bf a0 = *(const v8bf*)(Ar + kk + koff);
    v8bf a1 = *(const v8bf*)(Ar + kk + 16 + koff);
    v16bf a = __builtin_shufflevector(a0, a1, 0,1,2,3,4,5,6,7,8,9,10,11,12,13,14,15);
#pragma unroll
    for (int t = 0; t < 4; ++t) {
      const __bf16* Bc = BT + (size_t)(t * 16 + mr) * sB + kk;
      v8bf b0 = *(const v8bf*)(Bc + koff);
      v8bf b1 = *(const v8bf*)(Bc + 16 + koff);
      v16bf b = __builtin_shufflevector(b0, b1, 0,1,2,3,4,5,6,7,8,9,10,11,12,13,14,15);
      acc[t] = __builtin_amdgcn_wmma_f32_16x16x32_bf16(false, a, false, b,
                                                       (short)0, acc[t], false, false);
    }
  }
}

// Epilogue: D VGPR r holds (m = r + 8*(lane>>4), n = lane&15).
__device__ __forceinline__ void store_tile16x64(v8f acc[4], int tm, int tn, int ldc,
    const float* __restrict__ bias, float* __restrict__ Cf, __bf16* __restrict__ Cb,
    int act, int lane)
{
  const int half = lane >> 4;
  const int nr   = lane & 15;
#pragma unroll
  for (int t = 0; t < 4; ++t) {
    int n = tn + t * 16 + nr;
    float bn = bias ? bias[n] : 0.f;
#pragma unroll
    for (int r = 0; r < 8; ++r) {
      float v = acc[t][r] + bn;
      if (act) v = tanhf(v);
      size_t idx = (size_t)(tm + half * 8 + r) * ldc + n;
      if (Cf) Cf[idx] = v;
      if (Cb) Cb[idx] = (__bf16)v;
    }
  }
}

// ---------------- weight convert+transpose: WT[n*K+k] = (bf16)W[k*N+n]
__global__ void k_cvt_t(const float* __restrict__ W, __bf16* __restrict__ WT,
                        int K, int N)
{
  int i = blockIdx.x * 256 + threadIdx.x;
  if (i >= K * N) return;
  int n = i / K, k = i - n * K;
  WT[i] = (__bf16)W[(size_t)k * N + n];
}

// ---------------- bag-of-words gather+sum via async global->LDS staging
__global__ void k_bow(const int* __restrict__ sents, const float* __restrict__ emb,
                      __bf16* __restrict__ Xbf)
{
  __shared__ float rows[32 * DD];              // 32 KB staging (two passes)
  int bn = blockIdx.x;                          // 0..B*N-1
  int t  = threadIdx.x;                         // 0..255
  const int* s = sents + (size_t)bn * LL;
  float acc = 0.f;
  for (int half = 0; half < 2; ++half) {
#if defined(HAVE_ASYNC_LDS)
    // 32 rows x 256 f32 = 2048 16-byte segments; 8 per thread, ASYNCcnt-tracked
#pragma unroll
    for (int i = 0; i < 8; ++i) {
      int seg = t + 256 * i;                    // 0..2047
      int l   = seg >> 6;                       // row within batch of 32
      int c   = (seg & 63) << 2;                // float offset within row
      const float* src = emb + (size_t)s[half * 32 + l] * DD + c;
      __builtin_amdgcn_global_load_async_to_lds_b128(
          (as1_v4i)src, (as3_v4i)&rows[l * DD + c], 0, 0);
    }
    wait_asynccnt0();
    __syncthreads();
#pragma unroll 4
    for (int l = 0; l < 32; ++l) acc += rows[l * DD + t];
    __syncthreads();
#else
    for (int l = 0; l < 32; ++l)
      acc += emb[(size_t)s[half * 32 + l] * DD + t];
#endif
  }
  Xbf[(size_t)bn * DD + t] = (__bf16)acc;
}

// ---------------- title sum -> bf16
__global__ void k_title(const int* __restrict__ titles, const float* __restrict__ emb,
                        __bf16* __restrict__ Tsum)
{
  int b = blockIdx.x, d = threadIdx.x;
  const int* tt = titles + (size_t)b * NN * TT;  // titles[b,0,:]
  float acc = 0.f;
  for (int i = 0; i < TT; ++i)
    acc += emb[(size_t)tt[i] * DD + d];
  Tsum[b * DD + d] = (__bf16)acc;
}

// ---------------- generic GEMM: C[MxN] = A[MxK] @ BT[NxK]^T (+bias)(opt tanh)
__global__ void k_gemm_bf(const __bf16* __restrict__ A, const __bf16* __restrict__ BT,
                          const float* __restrict__ bias,
                          float* __restrict__ Cf, __bf16* __restrict__ Cb,
                          int M, int N, int K, int act)
{
  int lane = threadIdx.x & 31;
  int wave = threadIdx.x >> 5;
  int tile = blockIdx.x * (blockDim.x >> 5) + wave;
  int ng   = N >> 6;                                  // 64-wide column groups
  if (tile >= (M >> 4) * ng) return;                  // wave-uniform guard
  int tm = (tile / ng) << 4;
  int tn = (tile % ng) << 6;
  v8f acc[4] = {};
  wmma_tile16x64(acc, A + (size_t)tm * K, K, BT + (size_t)tn * K, K, K, lane);
  store_tile16x64(acc, tm, tn, N, bias, Cf, Cb, act, lane);
}

// ---------------- GRU gate fusion for step n (gi/gh already include biases)
__global__ void k_gru_gates(const float* __restrict__ GI, const float* __restrict__ GH,
                            float* __restrict__ Hcur, __bf16* __restrict__ Hbf,
                            __bf16* __restrict__ HSbf, __bf16* __restrict__ HSTbf, int n)
{
  int i = blockIdx.x * blockDim.x + threadIdx.x;     // 0..B*H-1
  if (i >= BB * HH) return;
  int b = i / HH, j = i % HH;
  const float* gi = GI + ((size_t)b * NN + n) * H3;
  const float* gh = GH + (size_t)b * H3;
  float r  = 1.f / (1.f + expf(-(gi[j] + gh[j])));
  float z  = 1.f / (1.f + expf(-(gi[HH + j] + gh[HH + j])));
  float nn = tanhf(gi[2 * HH + j] + r * gh[2 * HH + j]);
  float hv = (1.f - z) * nn + z * Hcur[i];
  Hcur[i] = hv;
  Hbf[i]  = (__bf16)hv;
  HSbf[((size_t)b * NN + n) * HH + j] = (__bf16)hv;   // row-major [b,n,h]
  HSTbf[((size_t)b * HH + j) * NN + n] = (__bf16)hv;  // transposed [b,h,n]
}

// ---------------- attention scores: AL[b] = Q[b] @ HS[b]^T (one 16x64 tile/wave)
__global__ void k_attn_scores(const __bf16* __restrict__ Qbf, const __bf16* __restrict__ HSbf,
                              float* __restrict__ AL)
{
  int lane = threadIdx.x & 31;
  int wave = threadIdx.x >> 5;
  int tile = blockIdx.x * 8 + wave;                   // 128 tiles: 32 b x 4 m-tiles
  int b  = tile >> 2;
  int tm = (tile & 3) << 4;
  v8f acc[4] = {};
  wmma_tile16x64(acc, Qbf + ((size_t)b * NN + tm) * HH, HH,
                 HSbf + (size_t)b * NN * HH, HH, HH, lane);
  store_tile16x64(acc, tm, 0, NN, nullptr, AL + (size_t)b * NN * NN, nullptr, 0, lane);
}

// ---------------- causal-masked softmax (one wave per row), writes bf16 probs
__global__ void k_softmax(const float* __restrict__ AL, __bf16* __restrict__ ALbf)
{
  int lane = threadIdx.x & 31;
  int wave = threadIdx.x >> 5;
  int row  = blockIdx.x * 8 + wave;                   // 0..2047 == b*N + q
  int q = row & (NN - 1);
  const float* R = AL + (size_t)row * NN;
  const float MASKLOG = -103.2789306640625f;          // logf((float)1e-45)
  float v0 = R[lane]      + ((lane      <= q) ? 0.f : MASKLOG);
  float v1 = R[lane + 32] + ((lane + 32 <= q) ? 0.f : MASKLOG);
  float m = fmaxf(v0, v1);
  for (int s = 16; s; s >>= 1) m = fmaxf(m, __shfl_xor(m, s, 32));
  float e0 = expf(v0 - m), e1 = expf(v1 - m);
  float sum = e0 + e1;
  for (int s = 16; s; s >>= 1) sum += __shfl_xor(sum, s, 32);
  float inv = 1.f / sum;
  ALbf[(size_t)row * NN + lane]      = (__bf16)(e0 * inv);
  ALbf[(size_t)row * NN + lane + 32] = (__bf16)(e1 * inv);
}

// ---------------- context: CTX[b] = a[b] @ HS[b]  (uses transposed HS copy)
__global__ void k_attn_ctx(const __bf16* __restrict__ ALbf, const __bf16* __restrict__ HSTbf,
                           __bf16* __restrict__ CTXbf)
{
  int lane = threadIdx.x & 31;
  int wave = threadIdx.x >> 5;
  int tile = blockIdx.x * 8 + wave;                   // 1024 tiles: 32 b x 4 m x 8 ngrp
  int b  = tile >> 5;
  int t  = tile & 31;
  int tm = (t >> 3) << 4;
  int tn = (t & 7) << 6;
  v8f acc[4] = {};
  wmma_tile16x64(acc, ALbf + (size_t)b * NN * NN + (size_t)tm * NN, NN,
                 HSTbf + (size_t)b * HH * NN + (size_t)tn * NN, NN, NN, lane);
  store_tile16x64(acc, tm, tn, HH, nullptr, nullptr, CTXbf + (size_t)b * NN * HH, 0, lane);
}

// ---------------- out = tanh(concat(CTX,HS) @ W_att_out + b): two K-passes
__global__ void k_gemm_cat(const __bf16* __restrict__ A1, const __bf16* __restrict__ A2,
                           const __bf16* __restrict__ WoT,   // [512 n][1024 k]
                           const float* __restrict__ bias, float* __restrict__ C)
{
  int lane = threadIdx.x & 31;
  int wave = threadIdx.x >> 5;
  int tile = blockIdx.x * 8 + wave;                   // 1024 tiles: M=2048, N=512
  int ng = HH >> 6;                                   // 8
  int tm = (tile / ng) << 4;
  int tn = (tile % ng) << 6;
  v8f acc[4] = {};
  wmma_tile16x64(acc, A1 + (size_t)tm * HH, HH, WoT + (size_t)tn * (2 * HH),       2 * HH, HH, lane);
  wmma_tile16x64(acc, A2 + (size_t)tm * HH, HH, WoT + (size_t)tn * (2 * HH) + HH,  2 * HH, HH, lane);
  store_tile16x64(acc, tm, tn, HH, bias, C, nullptr, 1, lane);
}

// ---------------------------------------------------------------------------
extern "C" void kernel_launch(void* const* d_in, const int* in_sizes, int n_in,
                              void* d_out, int out_size, void* d_ws, size_t ws_size,
                              hipStream_t stream)
{
  (void)in_sizes; (void)n_in; (void)out_size; (void)ws_size;
  const int*   sents     = (const int*)d_in[0];
  const int*   titles    = (const int*)d_in[1];
  const float* emb       = (const float*)d_in[2];
  const float* Wt        = (const float*)d_in[3];
  const float* bt        = (const float*)d_in[4];
  const float* W_ih      = (const float*)d_in[5];
  const float* W_hh      = (const float*)d_in[6];
  const float* b_ih      = (const float*)d_in[7];
  const float* b_hh      = (const float*)d_in[8];
  const float* W_att_in  = (const float*)d_in[9];
  const float* W_att_out = (const float*)d_in[10];
  const float* b_att_out = (const float*)d_in[11];
  float* out = (float*)d_out;

  // workspace carve-up (256B aligned per buffer)
  char* p = (char*)d_ws;
  auto alloc = [&](size_t bytes) { char* r = p; p += (bytes + 255) & ~(size_t)255; return r; };
  float*  GI    = (float*) alloc((size_t)BB * NN * H3 * 4);   // [2048,1536] f32
  float*  GH    = (float*) alloc((size_t)BB * H3 * 4);        // [32,1536]   f32
  float*  Hcur  = (float*) alloc((size_t)BB * HH * 4);        // [32,512]    f32
  float*  AL    = (float*) alloc((size_t)BB * NN * NN * 4);   // [32,64,64]  f32
  __bf16* Xbf   = (__bf16*)alloc((size_t)BB * NN * DD * 2);   // [2048,256]
  __bf16* Tsum  = (__bf16*)alloc((size_t)BB * DD * 2);        // [32,256]
  __bf16* Hbf   = (__bf16*)alloc((size_t)BB * HH * 2);        // [32,512]
  __bf16* HSbf  = (__bf16*)alloc((size_t)BB * NN * HH * 2);   // [2048,512]
  __bf16* HSTbf = (__bf16*)alloc((size_t)BB * HH * NN * 2);   // [32,512,64]
  __bf16* Qbf   = (__bf16*)alloc((size_t)BB * NN * HH * 2);   // [2048,512]
  __bf16* ALbf  = (__bf16*)alloc((size_t)BB * NN * NN * 2);   // [32,64,64]
  __bf16* CTXbf = (__bf16*)alloc((size_t)BB * NN * HH * 2);   // [2048,512]
  __bf16* WihT  = (__bf16*)alloc((size_t)H3 * DD * 2);        // [1536,256]
  __bf16* WhhT  = (__bf16*)alloc((size_t)H3 * HH * 2);        // [1536,512]
  __bf16* WtT   = (__bf16*)alloc((size_t)HH * DD * 2);        // [512,256]
  __bf16* WinT  = (__bf16*)alloc((size_t)HH * HH * 2);        // [512,512]
  __bf16* WoT   = (__bf16*)alloc((size_t)HH * 2 * HH * 2);    // [512,1024]

  // 0) weights -> bf16 transposed (k-contiguous for b128 operand loads)
  k_cvt_t<<<(DD * H3 + 255) / 256, 256, 0, stream>>>(W_ih,      WihT, DD, H3);
  k_cvt_t<<<(HH * H3 + 255) / 256, 256, 0, stream>>>(W_hh,      WhhT, HH, H3);
  k_cvt_t<<<(DD * HH + 255) / 256, 256, 0, stream>>>(Wt,        WtT,  DD, HH);
  k_cvt_t<<<(HH * HH + 255) / 256, 256, 0, stream>>>(W_att_in,  WinT, HH, HH);
  k_cvt_t<<<(2*HH*HH + 255) / 256, 256, 0, stream>>>(W_att_out, WoT,  2 * HH, HH);

  // 1) gathers
  k_bow  <<<BB * NN, DD, 0, stream>>>(sents, emb, Xbf);
  k_title<<<BB,      DD, 0, stream>>>(titles, emb, Tsum);

  // 2) hoisted GI = X @ W_ih + b_ih   [2048,256]x[256,1536]  (3072 wave-tiles)
  k_gemm_bf<<<3072 / 8, 256, 0, stream>>>(Xbf, WihT, b_ih, GI, nullptr,
                                          BB * NN, H3, DD, 0);
  // 3) h0 = Tsum @ Wt + bt            [32,256]x[256,512]     (16 wave-tiles)
  k_gemm_bf<<<2, 256, 0, stream>>>(Tsum, WtT, bt, Hcur, Hbf, BB, HH, DD, 0);

  // 4) sequential GRU: GH = h @ W_hh + b_hh, then fused gates (48 wave-tiles/step)
  for (int n = 0; n < NN; ++n) {
    k_gemm_bf<<<6, 256, 0, stream>>>(Hbf, WhhT, b_hh, GH, nullptr, BB, H3, HH, 0);
    k_gru_gates<<<(BB * HH) / 256, 256, 0, stream>>>(GI, GH, Hcur, Hbf, HSbf, HSTbf, n);
  }

  // 5) attention
  k_gemm_bf<<<1024 / 8, 256, 0, stream>>>(HSbf, WinT, nullptr, nullptr, Qbf,
                                          BB * NN, HH, HH, 0);          // Q
  k_attn_scores<<<128 / 8,  256, 0, stream>>>(Qbf, HSbf, AL);
  k_softmax    <<<2048 / 8, 256, 0, stream>>>(AL, ALbf);
  k_attn_ctx   <<<1024 / 8, 256, 0, stream>>>(ALbf, HSTbf, CTXbf);
  k_gemm_cat   <<<1024 / 8, 256, 0, stream>>>(CTXbf, HSbf, WoT, b_att_out, out);
}
